// MHN_76046690943124
// MI455X (gfx1250) — matrix operand order repaired and checked by reading, not compiled, not run on hardware
//
#include <hip/hip_runtime.h>
#include <hip/hip_bf16.h>
#include <math.h>

typedef __attribute__((ext_vector_type(16))) _Float16 v16h;
typedef __attribute__((ext_vector_type(8)))  _Float16 v8h;
typedef __attribute__((ext_vector_type(8)))  float    v8f;

// Problem constants
constexpr int Bn   = 16;
constexpr int INP  = 64;
constexpr int OUP  = 64;
constexpr int Td   = 30;
constexpr int Hd   = 64;
constexpr int Wd   = 44;
constexpr int Kk   = 27 * 64;          // 1728 reduction depth
constexpr int KSTEPS = Kk / 32;        // 54
constexpr int SPAT = Td * Hd * Wd;     // 84480
constexpr int LDS_W = 52;              // padded W (need up to wi=49)

static __device__ __forceinline__ float sigmoidf_(float x) {
    return 1.0f / (1.0f + __expf(-x));
}

// ---------------- Kernel 1: global average pool ----------------
// one block per (b, ic); 1024 blocks
__global__ __launch_bounds__(256) void gap_kernel(const float* __restrict__ x,
                                                  float* __restrict__ xgap) {
    const int bc  = blockIdx.x;            // b*64 + ic
    const int tid = threadIdx.x;
    const float* p = x + (size_t)bc * SPAT;
    float s = 0.f;
    for (int i = tid; i < SPAT; i += 256) s += p[i];
    __shared__ float red[256];
    red[tid] = s;
    __syncthreads();
    for (int k = 128; k > 0; k >>= 1) {
        if (tid < k) red[tid] += red[tid + k];
        __syncthreads();
    }
    if (tid == 0) xgap[bc] = red[0] * (1.0f / (float)SPAT);
}

// ---------------- Kernel 2: tiny FCs (reduce + fc1 + sigmoid) ----------------
// single block, 256 threads
__global__ __launch_bounds__(256) void fc_kernel(const float* __restrict__ xgap,
                                                 const float* __restrict__ w_reduce,
                                                 const float* __restrict__ b_reduce,
                                                 const float* __restrict__ w_fc1,
                                                 const float* __restrict__ b_fc1,
                                                 float* __restrict__ hout) {
    __shared__ float gs[16 * 16];          // g[b][j]
    const int tid = threadIdx.x;
    if (tid < 256) {                       // 16*16 entries
        const int b = tid >> 4, j = tid & 15;
        float s = b_reduce[j];
        #pragma unroll
        for (int c = 0; c < 64; ++c) s += xgap[b * 64 + c] * w_reduce[j * 64 + c];
        gs[tid] = s;
    }
    __syncthreads();
    for (int e = tid; e < Bn * 128; e += 256) {   // h[b][m], m<128
        const int b = e >> 7, m = e & 127;
        float s = b_fc1[m];
        #pragma unroll
        for (int j = 0; j < 16; ++j) s += gs[b * 16 + j] * w_fc1[m * 16 + j];
        hout[e] = sigmoidf_(s);
    }
}

// ---------------- Kernel 3: dynamic weight generation ----------------
// wdyn[b][oc][k'] (f16), k' = off3*64 + ic, off3 = kd*9+kh*3+kw
__global__ __launch_bounds__(256) void wgen_kernel(const float* __restrict__ w_fc2,
                                                   const float* __restrict__ h,
                                                   _Float16* __restrict__ wdyn) {
    const int idx = blockIdx.x * 256 + threadIdx.x;     // < 16*64*1728
    const int b    = idx / (OUP * Kk);
    const int r    = idx % (OUP * Kk);
    const int oc   = r / Kk;
    const int k    = r % Kk;
    const int off3 = k >> 6;          // 0..26
    const int ic   = k & 63;
    const int g    = 2 * oc + (ic >> 5);
    const int o    = (ic & 31) * 27 + off3;
    const float v  = w_fc2[g * 864 + o] * h[b * 128 + g];
    wdyn[idx] = (_Float16)v;
}

// ---------------- Kernel 4: implicit-GEMM dynamic conv3d via WMMA ----------------
// one workgroup per (b, t, h); 128 threads = 4 waves; wave = oc tile (M=16)
__global__ __launch_bounds__(128) void conv_kernel(const float* __restrict__ x,
                                                   const _Float16* __restrict__ wdyn,
                                                   float* __restrict__ out) {
    __shared__ _Float16 xs[9 * LDS_W * 64];    // [kd*3+kh][wi][ic], 59904 B

    const int bid = blockIdx.x;
    const int hh  = bid % Hd;
    const int tt  = (bid / Hd) % Td;
    const int b   = bid / (Hd * Td);
    const int tid = threadIdx.x;

    // ---- stage input slab into LDS (f16, ic innermost) ----
    for (int e = tid; e < 9 * 64 * LDS_W; e += 128) {
        const int wi  = e % LDS_W;
        const int t2  = e / LDS_W;
        const int ic  = t2 & 63;
        const int off = t2 >> 6;               // kd*3+kh
        const int kd  = off / 3, kh = off % 3;
        const int ti  = tt + kd - 1;
        const int hi  = hh + kh - 1;
        const int wIn = wi - 1;
        float v = 0.f;
        if ((unsigned)ti < (unsigned)Td && (unsigned)hi < (unsigned)Hd &&
            (unsigned)wIn < (unsigned)Wd) {
            v = x[((((size_t)b * INP + ic) * Td + ti) * Hd + hi) * Wd + wIn];
        }
        xs[(off * LDS_W + wi) * 64 + ic] = (_Float16)v;
    }
    __syncthreads();

    // ---- WMMA main loop ----
    const int lane = tid & 31;
    const int wave = tid >> 5;                 // m-tile
    const int nlo  = lane & 15;                // A row / B column / output w-in-tile
    const bool hiH = lane >= 16;
    const int kbA  = hiH ? 16 : 0;             // A: K in {kbA..kbA+15} contiguous
    const int kbB  = hiH ? 16 : 0;             // B: K in {kbB..kbB+15} contiguous

    const _Float16* Ap = wdyn + ((size_t)(b * OUP + wave * 16 + nlo)) * Kk;

    v8f c0 = {}, c1 = {}, c2 = {};

    for (int s = 0; s < KSTEPS; ++s) {
        // A fragment: two contiguous 16B runs
        const _Float16* ap = Ap + s * 32 + kbA;
        v8h alo = *(const v8h*)ap;
        v8h ahi = *(const v8h*)(ap + 8);
        v16h a = __builtin_shufflevector(alo, ahi,
                 0,1,2,3,4,5,6,7,8,9,10,11,12,13,14,15);

        const int off3 = s >> 1;               // kd*9+kh*3+kw
        const int ldo  = off3 / 3;             // kd*3+kh  (LDS row-pair)
        const int kw   = off3 % 3;
        const int icB  = ((s & 1) << 5) + kbB; // base ic for this lane's K run
        const _Float16* Bb = xs + (ldo * LDS_W + kw) * 64 + icB;

        // n-tile 0 (w = 0..15)
        {
            const _Float16* p = Bb + (0 * 16 + nlo) * 64;
            v8h blo = *(const v8h*)p;
            v8h bhi = *(const v8h*)(p + 8);
            v16h bv = __builtin_shufflevector(blo, bhi,
                      0,1,2,3,4,5,6,7,8,9,10,11,12,13,14,15);
            c0 = __builtin_amdgcn_wmma_f32_16x16x32_f16(false, a, false, bv,
                                                        (short)0, c0, false, false);
        }
        // n-tile 1 (w = 16..31)
        {
            const _Float16* p = Bb + (1 * 16 + nlo) * 64;
            v8h blo = *(const v8h*)p;
            v8h bhi = *(const v8h*)(p + 8);
            v16h bv = __builtin_shufflevector(blo, bhi,
                      0,1,2,3,4,5,6,7,8,9,10,11,12,13,14,15);
            c1 = __builtin_amdgcn_wmma_f32_16x16x32_f16(false, a, false, bv,
                                                        (short)0, c1, false, false);
        }
        // n-tile 2 (w = 32..47, store only w < 44)
        {
            const _Float16* p = Bb + (2 * 16 + nlo) * 64;
            v8h blo = *(const v8h*)p;
            v8h bhi = *(const v8h*)(p + 8);
            v16h bv = __builtin_shufflevector(blo, bhi,
                      0,1,2,3,4,5,6,7,8,9,10,11,12,13,14,15);
            c2 = __builtin_amdgcn_wmma_f32_16x16x32_f16(false, a, false, bv,
                                                        (short)0, c2, false, false);
        }
    }

    // ---- store C tiles: VGPR r -> M = r (lanes 0-15) / r+8 (lanes 16-31) ----
    #pragma unroll
    for (int r = 0; r < 8; ++r) {
        const int oc = wave * 16 + r + (hiH ? 8 : 0);
        const size_t o = ((((size_t)b * OUP + oc) * Td + tt) * Hd + hh) * Wd;
        out[o + nlo]      = c0[r];             // w = 0..15
        out[o + 16 + nlo] = c1[r];             // w = 16..31
        if (nlo < 12) out[o + 32 + nlo] = c2[r];   // w = 32..43
    }
}

// ---------------- launch ----------------
extern "C" void kernel_launch(void* const* d_in, const int* in_sizes, int n_in,
                              void* d_out, int out_size, void* d_ws, size_t ws_size,
                              hipStream_t stream) {
    const float* x        = (const float*)d_in[0];
    const float* w_reduce = (const float*)d_in[1];
    const float* b_reduce = (const float*)d_in[2];
    const float* w_fc1    = (const float*)d_in[3];
    const float* b_fc1    = (const float*)d_in[4];
    const float* w_fc2    = (const float*)d_in[5];
    float* out = (float*)d_out;

    // workspace layout
    float*    xgap = (float*)d_ws;                         // 1024 floats
    float*    hbuf = xgap + Bn * INP;                      // 2048 floats
    _Float16* wdyn = (_Float16*)((char*)d_ws + 4 * (Bn * INP + Bn * 128)); // 16B aligned

    gap_kernel<<<Bn * INP, 256, 0, stream>>>(x, xgap);
    fc_kernel<<<1, 256, 0, stream>>>(xgap, w_reduce, b_reduce, w_fc1, b_fc1, hbuf);
    wgen_kernel<<<(Bn * OUP * Kk) / 256, 256, 0, stream>>>(w_fc2, hbuf, wdyn);
    conv_kernel<<<Bn * Td * Hd, 128, 0, stream>>>(x, wdyn, out);
}
